// Get_embd_11836929868522
// MI455X (gfx1250) — matrix operand order repaired and checked by reading, not compiled, not run on hardware
//
#include <hip/hip_runtime.h>

// ---------------------------------------------------------------------------
// Types & helpers for CDNA5 WMMA (wave32, v_wmma_f32_16x16x32_bf16)
// ---------------------------------------------------------------------------
typedef __attribute__((ext_vector_type(16))) __bf16 v16bf;
typedef __attribute__((ext_vector_type(8)))  float  v8f;

union V16U { v16bf v; uint4 u[2]; __bf16 e[16]; };
union V8U  { v8f v; float f[8]; };

__device__ __forceinline__ __bf16 f2bf(float x) {
    unsigned u = __builtin_bit_cast(unsigned, x);
    unsigned r = (u + 0x7FFFu + ((u >> 16) & 1u)) >> 16;   // round-to-nearest-even
    unsigned short h = (unsigned short)r;
    return __builtin_bit_cast(__bf16, h);
}

__device__ __forceinline__ v8f wmma_bf(v16bf a, v16bf b, v8f c) {
    return __builtin_amdgcn_wmma_f32_16x16x32_bf16(false, a, false, b, (short)0, c, false, false);
}

// A operand (16x32 bf16), row-major source. lane 0-15: row=lane, K k0..k0+15;
// lanes 16-31: row=lane-16, K k0+16..k0+31.
__device__ __forceinline__ v16bf ldA(const __bf16* base, int stride, int row0, int k0, int lane) {
    V16U r;
    int row = row0 + (lane & 15);
    int kb  = k0 + ((lane >> 4) << 4);
    const __bf16* p = base + row * stride + kb;
    r.u[0] = *(const uint4*)(p);
    r.u[1] = *(const uint4*)(p + 8);
    return r.v;
}

// B operand (32x16 bf16) where B[k][n] = W[n][k], W row-major (stride elements).
// lane 0-15: col=n0+lane, K k0..k0+15 ; lanes 16-31: K k0+16..k0+31. Contiguous 16 elems.
__device__ __forceinline__ v16bf ldBw(const __bf16* w, int stride, int n0, int k0, int lane) {
    V16U r;
    int n  = n0 + (lane & 15);
    int kb = k0 + ((lane >> 4) << 4);
    const __bf16* p = w + n * stride + kb;
    r.u[0] = *(const uint4*)(p);
    r.u[1] = *(const uint4*)(p + 8);
    return r.v;
}

// ---------------------------------------------------------------------------
// f32 -> bf16 conversion
// ---------------------------------------------------------------------------
__global__ __launch_bounds__(256) void cvt_kernel(const float* __restrict__ s,
                                                  __bf16* __restrict__ d, int n) {
    int i = blockIdx.x * 256 + threadIdx.x;
    if (i < n) d[i] = f2bf(s[i]);
}

// ---------------------------------------------------------------------------
// Fused embedding-gather + QKV projection + scrambled-view attention.
// One block per attention group (n = 3b+j, s'). Grid: (50, 96), block 256.
// Writes temp (already laid out for the final reshape) as bf16 [32][38400].
// Dynamic LDS: 89476 bytes.
// ---------------------------------------------------------------------------
__global__ __launch_bounds__(256)
void fused_qkv_attn(const int*   __restrict__ codes,      // [32,50,32]
                    const float* __restrict__ mask_code,  // [32,50,32,1]
                    const float* __restrict__ emb,        // [10001,256]
                    const __bf16* __restrict__ wcat,      // [2304,256] bf16 (q,k,v rows)
                    const float* __restrict__ bq,
                    const float* __restrict__ bk,
                    const float* __restrict__ bv,
                    __bf16* __restrict__ temp)            // [32][38400] bf16
{
    extern __shared__ char sm[];
    __bf16* curA    = (__bf16*)(sm);          // [16][264]
    __bf16* proj    = (__bf16*)(sm + 8448);   // [16][2304]  cols: 0..767 Q, 768.. K, 1536.. V
    float*  attf    = (float*) (sm + 82176);  // [32][32]
    __bf16* attb    = (__bf16*)(sm + 86272);  // [32][32]
    float*  nom     = (float*) (sm + 88320);  // [256]
    int*    shc     = (int*)   (sm + 89344);  // count
    int*    rowcode = (int*)   (sm + 89348);  // [16]
    float*  rowmask = (float*) (sm + 89412);  // [16]

    const int sp = blockIdx.x;            // s' in [0,50)
    const int n  = blockIdx.y;            // in [0,96)
    const int b  = n / 3;
    const int j  = n % 3;
    const int bm = n & 31;                // faithful n % B mask indexing
    const int a  = j * 409600 + sp * 8192;  // flat base of the group slice
    const int m0 = a / 768;               // first projection row covered

    const int tid  = threadIdx.x;
    const int lane = tid & 31;
    const int wv   = tid >> 5;

    if (tid < 16) {
        int m = m0 + tid;
        if (m < 1600) {
            int s = m >> 5, c = m & 31;
            rowcode[tid] = codes[(b * 50 + s) * 32 + c];
            rowmask[tid] = mask_code[(b * 50 + s) * 32 + c];
        } else {
            rowcode[tid] = 0; rowmask[tid] = 0.f;
        }
    }
    __syncthreads();

    // embedding gather * mask_code -> bf16 A tile [16][256] (stride 264)
    for (int r = 0; r < 16; ++r) {
        float e = emb[rowcode[r] * 256 + tid] * rowmask[r];
        curA[r * 264 + tid] = f2bf(e);
    }
    __syncthreads();

    // projection GEMM: [16 x 2304] = curA[16x256] @ Wcat^T, K=256 (8 ksteps), 144 ntiles
    for (int nt = wv; nt < 144; nt += 8) {
        V8U acc;
        #pragma unroll
        for (int i = 0; i < 8; ++i) acc.f[i] = 0.f;
        for (int ks = 0; ks < 8; ++ks) {
            v16bf av = ldA(curA, 264, 0, ks * 32, lane);
            v16bf bw = ldBw(wcat, 256, nt * 16, ks * 32, lane);
            acc.v = wmma_bf(av, bw, acc.v);
        }
        int col = nt * 16 + (lane & 15);
        float bias = (col < 768) ? bq[col] : ((col < 1536) ? bk[col - 768] : bv[col - 1536]);
        int rbase = (lane >> 4) << 3;
        #pragma unroll
        for (int r = 0; r < 8; ++r)
            proj[(rbase + r) * 2304 + col] = f2bf(acc.f[r] + bias);
    }
    __syncthreads();

    // att = (q @ k^T) * scale   (scale = (256//3)**-0.5, faithful)
    const float scale = 0.10846522890932808f;
    if (wv < 4) {
        int mi = wv >> 1, ti = wv & 1;
        V8U acc;
        #pragma unroll
        for (int i = 0; i < 8; ++i) acc.f[i] = 0.f;
        for (int ks = 0; ks < 8; ++ks) {
            // q A-operand via flat map (Q section)
            V16U av;
            {
                int cq = mi * 16 + (lane & 15);
                int d0 = ks * 32 + ((lane >> 4) << 4);
                int f0 = a + cq * 256 + d0;
                int m_ = f0 / 768, t_ = f0 - m_ * 768;
                av.u[0] = *(const uint4*)(proj + (m_ - m0) * 2304 + t_);
                int f1 = f0 + 8;
                m_ = f1 / 768; t_ = f1 - m_ * 768;
                av.u[1] = *(const uint4*)(proj + (m_ - m0) * 2304 + t_);
            }
            // k B-operand via flat map (K section, +768)
            V16U bw;
            {
                int tk = ti * 16 + (lane & 15);
                int d0 = ks * 32 + ((lane >> 4) << 4);
                int f0 = a + tk * 256 + d0;
                int m_ = f0 / 768, t_ = f0 - m_ * 768;
                const __bf16* p = proj + (m_ - m0) * 2304 + 768 + t_;
                bw.u[0] = *(const uint4*)(p);
                bw.u[1] = *(const uint4*)(p + 8);
            }
            acc.v = wmma_bf(av.v, bw.v, acc.v);
        }
        int tk = ti * 16 + (lane & 15);
        int rbase = (lane >> 4) << 3;
        #pragma unroll
        for (int r = 0; r < 8; ++r)
            attf[(mi * 16 + rbase + r) * 32 + tk] = acc.f[r] * scale;
    }
    __syncthreads();

    // softmax over keys with mask_code[bm] (key axis); rows owned by threads 0..31
    if (tid < 32) {
        int cq = tid;
        float maxv = -3.0e38f; int cnt = 0;
        for (int t = 0; t < 32; ++t) {
            float mcv = mask_code[(bm * 50 + sp) * 32 + t];
            if (mcv != 0.f) { ++cnt; float v = attf[cq * 32 + t]; if (v > maxv) maxv = v; }
        }
        if (cnt > 0) {
            float sum = 0.f;
            for (int t = 0; t < 32; ++t) {
                float mcv = mask_code[(bm * 50 + sp) * 32 + t];
                float e = (mcv != 0.f) ? __expf(attf[cq * 32 + t] - maxv) : 0.f;
                attf[cq * 32 + t] = e; sum += e;
            }
            float inv = 1.f / sum;
            for (int t = 0; t < 32; ++t) attb[cq * 32 + t] = f2bf(attf[cq * 32 + t] * inv);
        } else {
            for (int t = 0; t < 32; ++t) attb[cq * 32 + t] = f2bf(0.f);
        }
        if (tid == 0) shc[0] = cnt;
    }
    __syncthreads();

    // tmp = att @ v, query-mask rows, column-sum -> nom[256]. Wave wv owns ni in {wv, wv+8}.
    for (int nsel = 0; nsel < 2; ++nsel) {
        int ni = wv + nsel * 8;
        // v B-operand (V section, +1536): 16 strided elements, shared by both mi tiles
        V16U bw;
        {
            int d  = ni * 16 + (lane & 15);
            int kb = (lane >> 4) << 4;
            #pragma unroll
            for (int i = 0; i < 16; ++i) {
                int f = a + (kb + i) * 256 + d;
                int m_ = f / 768, t_ = f - m_ * 768;
                bw.e[i] = proj[(m_ - m0) * 2304 + 1536 + t_];
            }
        }
        float partial = 0.f;
        for (int mi = 0; mi < 2; ++mi) {
            V8U acc;
            #pragma unroll
            for (int i = 0; i < 8; ++i) acc.f[i] = 0.f;
            v16bf av = ldA(attb, 32, mi * 16, 0, lane);   // K = 32, single step
            acc.v = wmma_bf(av, bw.v, acc.v);
            int rbase = (lane >> 4) << 3;
            #pragma unroll
            for (int r = 0; r < 8; ++r) {
                int cq = mi * 16 + rbase + r;
                partial += acc.f[r] * mask_code[(bm * 50 + sp) * 32 + cq];
            }
        }
        partial += __shfl_xor(partial, 16);               // combine lane halves (same column)
        if (lane < 16) nom[ni * 16 + lane] = partial;
    }
    __syncthreads();

    // temp write (already in post-reshape layout): temp[b][j*12800 + s'*256 + d]
    {
        int cnt = shc[0];
        float val = (cnt > 0) ? nom[tid] / (float)cnt : 0.f;
        temp[b * 38400 + j * 12800 + sp * 256 + tid] = f2bf(val);
    }
}

// ---------------------------------------------------------------------------
// Generic bf16 GEMM: C[M,N] = A[M,K] @ W^T + bias, A/W row-major bf16, C f32.
// One 16x16 tile per wave. Used for GI = temp @ gwih^T + gbih.
// ---------------------------------------------------------------------------
__global__ __launch_bounds__(256)
void gemm_bf16(const __bf16* __restrict__ A, const __bf16* __restrict__ W,
               const float* __restrict__ bias, float* __restrict__ C,
               int M, int N, int K, int lda, int ldb, int ldc)
{
    int tid = threadIdx.x, lane = tid & 31, wv = tid >> 5;
    int ntiles = N >> 4;
    int gid = blockIdx.x * 8 + wv;
    int total = (M >> 4) * ntiles;
    if (gid >= total) return;
    int mt = gid / ntiles;
    int nt = gid - mt * ntiles;

    V8U acc;
    #pragma unroll
    for (int i = 0; i < 8; ++i) acc.f[i] = 0.f;
    int ksteps = K >> 5;
    for (int ks = 0; ks < ksteps; ++ks) {
        v16bf av = ldA(A, lda, mt * 16, ks * 32, lane);
        v16bf bw = ldBw(W, ldb, nt * 16, ks * 32, lane);
        acc.v = wmma_bf(av, bw, acc.v);
    }
    int col = nt * 16 + (lane & 15);
    float bb = bias ? bias[col] : 0.f;
    int rbase = (lane >> 4) << 3;
    #pragma unroll
    for (int r = 0; r < 8; ++r)
        C[(mt * 16 + rbase + r) * ldc + col] = acc.f[r] + bb;
}

// ---------------------------------------------------------------------------
// Sequential bidirectional GRU. grid = 2 (blockIdx.x = direction), block 256.
// Dynamic LDS: 147968 bytes (h f32 32x256, gh f32 32x768, hbf bf16 32x264).
// ---------------------------------------------------------------------------
__global__ __launch_bounds__(256)
void gru_kernel(const float* __restrict__ GIf, const float* __restrict__ GIb,
                const __bf16* __restrict__ whhf, const __bf16* __restrict__ whhb,
                const float* __restrict__ bhhf, const float* __restrict__ bhhb,
                float* __restrict__ out)                      // [32][50][512]
{
    extern __shared__ char sm[];
    float*  h   = (float*)(sm);               // [32*256]
    float*  gh  = (float*)(sm + 32768);       // [32*768]
    __bf16* hbf = (__bf16*)(sm + 131072);     // [32][264]

    const int dir = blockIdx.x;
    const float*  GI  = dir ? GIb  : GIf;
    const __bf16* whh = dir ? whhb : whhf;
    const float*  bhh = dir ? bhhb : bhhf;

    const int tid = threadIdx.x, lane = tid & 31, wv = tid >> 5;

    for (int i = tid; i < 32 * 256; i += 256) h[i] = 0.f;
    __syncthreads();

    for (int ts = 0; ts < 50; ++ts) {
        int t = dir ? (49 - ts) : ts;
        // h -> bf16
        for (int i = tid; i < 32 * 256; i += 256) {
            int bb = i >> 8, u = i & 255;
            hbf[bb * 264 + u] = f2bf(h[i]);
        }
        __syncthreads();
        // gh = h @ whh^T : M=32 (2 mtiles), N=768 (48 ntiles), K=256
        for (int tile = wv; tile < 96; tile += 8) {
            int mt = tile / 48, nt = tile % 48;
            V8U acc;
            #pragma unroll
            for (int i = 0; i < 8; ++i) acc.f[i] = 0.f;
            for (int ks = 0; ks < 8; ++ks) {
                v16bf av = ldA(hbf, 264, mt * 16, ks * 32, lane);
                v16bf bw = ldBw(whh, 256, nt * 16, ks * 32, lane);
                acc.v = wmma_bf(av, bw, acc.v);
            }
            int col = nt * 16 + (lane & 15);
            int rbase = (lane >> 4) << 3;
            #pragma unroll
            for (int r = 0; r < 8; ++r)
                gh[(mt * 16 + rbase + r) * 768 + col] = acc.f[r];
        }
        __syncthreads();
        // gates + state update
        for (int i = tid; i < 32 * 256; i += 256) {
            int bb = i >> 8, u = i & 255;
            const float* gi = GI + (bb * 50 + t) * 768;
            float ir = gi[u], iz = gi[256 + u], in = gi[512 + u];
            float hr = gh[bb * 768 + u]        + bhh[u];
            float hz = gh[bb * 768 + 256 + u]  + bhh[256 + u];
            float hn = gh[bb * 768 + 512 + u]  + bhh[512 + u];
            float rr = 1.f / (1.f + __expf(-(ir + hr)));
            float zz = 1.f / (1.f + __expf(-(iz + hz)));
            float nn = tanhf(in + rr * hn);
            float hnew = (1.f - zz) * nn + zz * h[i];
            h[i] = hnew;
            out[(bb * 50 + t) * 512 + dir * 256 + u] = hnew;
        }
        __syncthreads();
    }
}

// ---------------------------------------------------------------------------
// us[b,t] = us_b + sum_d us_w[d] * tanh(ws_b[d] + ws_w[d,:] . out[b,t,:])
// ---------------------------------------------------------------------------
__global__ __launch_bounds__(256)
void score_kernel(const float* __restrict__ gout, const float* __restrict__ wsw,
                  const float* __restrict__ wsb, const float* __restrict__ usw,
                  const float* __restrict__ usb, float* __restrict__ us)
{
    __shared__ float red[256];
    int bs = blockIdx.x;                 // 0..1599
    int d  = threadIdx.x;
    const float* row = gout + (long)bs * 512;
    const float* wr  = wsw + (long)d * 512;
    float acc = wsb[d];
    for (int e = 0; e < 512; ++e) acc += wr[e] * row[e];
    red[d] = tanhf(acc) * usw[d];
    __syncthreads();
    for (int off = 128; off > 0; off >>= 1) {
        if (d < off) red[d] += red[d + off];
        __syncthreads();
    }
    if (d == 0) us[bs] = red[0] + usb[0];
}

// ---------------------------------------------------------------------------
// alpha = softmax over valid visits (via lengths); d_out[b,:] = sum_t alpha*out
// ---------------------------------------------------------------------------
__global__ __launch_bounds__(256)
void pool_kernel(const float* __restrict__ us, const int* __restrict__ lengths,
                 const float* __restrict__ gout, float* __restrict__ outp)
{
    __shared__ float alpha[64];
    int b = blockIdx.x;
    if (threadIdx.x == 0) {
        int L = lengths[b];
        float mx = -3.0e38f;
        for (int t = 0; t < 50; ++t) if (t < L && us[b * 50 + t] > mx) mx = us[b * 50 + t];
        float sum = 0.f;
        for (int t = 0; t < 50; ++t) {
            float e = (t < L) ? __expf(us[b * 50 + t] - mx) : 0.f;
            alpha[t] = e; sum += e;
        }
        float inv = 1.f / sum;
        for (int t = 0; t < 50; ++t) alpha[t] *= inv;
    }
    __syncthreads();
    for (int d = threadIdx.x; d < 512; d += 256) {
        float acc = 0.f;
        for (int t = 0; t < 50; ++t) acc += alpha[t] * gout[(b * 50 + t) * 512 + d];
        outp[b * 512 + d] = acc;
    }
}

// ---------------------------------------------------------------------------
// Host launcher
// ---------------------------------------------------------------------------
extern "C" void kernel_launch(void* const* d_in, const int* in_sizes, int n_in,
                              void* d_out, int out_size, void* d_ws, size_t ws_size,
                              hipStream_t stream) {
    (void)in_sizes; (void)n_in; (void)out_size; (void)ws_size;

    const int*   codes     = (const int*)  d_in[0];
    const float* mask_code = (const float*)d_in[4];
    const int*   lengths   = (const int*)  d_in[5];
    const float* emb       = (const float*)d_in[6];
    const float* wq_w = (const float*)d_in[7],  *wq_b = (const float*)d_in[8];
    const float* wk_w = (const float*)d_in[9],  *wk_b = (const float*)d_in[10];
    const float* wv_w = (const float*)d_in[11], *wv_b = (const float*)d_in[12];
    const float* gwih_f = (const float*)d_in[13], *gwhh_f = (const float*)d_in[14];
    const float* gbih_f = (const float*)d_in[15], *gbhh_f = (const float*)d_in[16];
    const float* gwih_b = (const float*)d_in[17], *gwhh_b = (const float*)d_in[18];
    const float* gbih_b = (const float*)d_in[19], *gbhh_b = (const float*)d_in[20];
    const float* ws_w = (const float*)d_in[21], *ws_b = (const float*)d_in[22];
    const float* us_w = (const float*)d_in[23], *us_b = (const float*)d_in[24];

    // workspace layout (all offsets 16B aligned)
    char* ws = (char*)d_ws;
    __bf16* wcat    = (__bf16*)ws;                 ws += (size_t)2304 * 256 * 2;
    __bf16* wihf_bf = (__bf16*)ws;                 ws += (size_t)768 * 768 * 2;
    __bf16* wihb_bf = (__bf16*)ws;                 ws += (size_t)768 * 768 * 2;
    __bf16* whhf_bf = (__bf16*)ws;                 ws += (size_t)768 * 256 * 2;
    __bf16* whhb_bf = (__bf16*)ws;                 ws += (size_t)768 * 256 * 2;
    __bf16* temp_bf = (__bf16*)ws;                 ws += (size_t)32 * 38400 * 2;
    float*  GIf     = (float*)ws;                  ws += (size_t)1600 * 768 * 4;
    float*  GIb     = (float*)ws;                  ws += (size_t)1600 * 768 * 4;
    float*  gruout  = (float*)ws;                  ws += (size_t)32 * 50 * 512 * 4;
    float*  usbuf   = (float*)ws;                  ws += (size_t)1600 * 4;

    const int SM1 = 89476;    // fused_qkv_attn dynamic LDS
    const int SM2 = 147968;   // gru_kernel dynamic LDS (CDNA5 WGP has 320 KB)
    (void)hipFuncSetAttribute(reinterpret_cast<const void*>(&fused_qkv_attn),
                              hipFuncAttributeMaxDynamicSharedMemorySize, SM1);
    (void)hipFuncSetAttribute(reinterpret_cast<const void*>(&gru_kernel),
                              hipFuncAttributeMaxDynamicSharedMemorySize, SM2);

    // weight conversions f32 -> bf16
    auto cvt = [&](const float* s, __bf16* d, int n) {
        cvt_kernel<<<(n + 255) / 256, 256, 0, stream>>>(s, d, n);
    };
    cvt(wq_w, wcat,                 768 * 256);
    cvt(wk_w, wcat + 768 * 256,     768 * 256);
    cvt(wv_w, wcat + 1536 * 256,    768 * 256);
    cvt(gwih_f, wihf_bf, 768 * 768);
    cvt(gwih_b, wihb_bf, 768 * 768);
    cvt(gwhh_f, whhf_bf, 768 * 256);
    cvt(gwhh_b, whhb_bf, 768 * 256);

    // fused embedding + QKV projection + attention -> temp
    fused_qkv_attn<<<dim3(50, 96), 256, SM1, stream>>>(
        codes, mask_code, emb, wcat, wq_b, wk_b, wv_b, temp_bf);

    // GI (time-parallel GRU input projections): [1600,768] @ [768,768]^T
    gemm_bf16<<<600, 256, 0, stream>>>(temp_bf, wihf_bf, gbih_f, GIf,
                                       1600, 768, 768, 768, 768, 768);
    gemm_bf16<<<600, 256, 0, stream>>>(temp_bf, wihb_bf, gbih_b, GIb,
                                       1600, 768, 768, 768, 768, 768);

    // sequential bidirectional GRU
    gru_kernel<<<2, 256, SM2, stream>>>(GIf, GIb, whhf_bf, whhb_bf,
                                        gbhh_f, gbhh_b, gruout);

    // attention pooling
    score_kernel<<<1600, 256, 0, stream>>>(gruout, ws_w, ws_b, us_w, us_b, usbuf);
    pool_kernel<<<32, 256, 0, stream>>>(usbuf, lengths, gruout, (float*)d_out);
}
